// SelectiveGatedElmanCell_2147483648054
// MI455X (gfx1250) — matrix operand order repaired and checked by reading, not compiled, not run on hardware
//
#include <hip/hip_runtime.h>
#include <hip/hip_bf16.h>

#define D    1024
#define Tn   2048
#define Bn   16
#define NBLK 16
#define SCAN_THREADS 128

typedef __attribute__((ext_vector_type(16))) __bf16 v16bf;
typedef __attribute__((ext_vector_type(8)))  float  v8f;
typedef unsigned short u16;

// ---------------- bf16 helpers ----------------
__device__ __forceinline__ u16 f2b(float f) {
  __bf16 h = (__bf16)f;
  return __builtin_bit_cast(u16, h);
}

// ---------------- WMMA fragment loaders (row-major bf16 buffers) ----------------
// A: 16xK tile, element (m,k). Lane l: m=l%16; lanes<16 hold K {0..7,16..23}+kb,
// lanes>=16 hold K {8..15,24..31}+kb  (ISA 7.12.2, 16-bit A layout).
__device__ __forceinline__ v16bf load_frag_A(const u16* base, int lane, int kb, int ld) {
  int m    = lane & 15;
  int half = lane >> 4;
  const char* p = (const char*)(base + (size_t)m * ld + kb + half * 8);
  union { v16bf v; uint4 q[2]; } f;
  f.q[0] = *(const uint4*)(p);
  f.q[1] = *(const uint4*)(p + 32);   // +16 elements
  return f.v;
}
// B: KxN tile where B[k][n] = W[n][k] (row-major W, rows are output columns).
// Lane l: n=n0+l%16; lanes<16 hold K {0..7,16..23}+kb, lanes>=16 hold K {8..15,24..31}+kb.
__device__ __forceinline__ v16bf load_frag_B(const u16* W, int lane, int n0, int kb, int ld) {
  int n = n0 + (lane & 15);
  int k = kb + ((lane >> 4) << 3);
  const char* p = (const char*)(W + (size_t)n * ld + k);
  union { v16bf v; uint4 q[2]; } f;
  f.q[0] = *(const uint4*)(p);
  f.q[1] = *(const uint4*)(p + 32);
  return f.v;
}

// ---------------- async LDS copy (gfx1250 GLOBAL_LOAD_ASYNC_TO_LDS) ----------------
// Copies 16 bytes from global to LDS; tracked on ASYNCcnt, no VGPR data movement.
__device__ __forceinline__ void async_b128_to_lds(unsigned lds_byte_off,
                                                  const void* gptr) {
  asm volatile("global_load_async_to_lds_b128 %0, %1, off"
               :: "v"(lds_byte_off),
                  "v"((unsigned long long)(uintptr_t)gptr)
               : "memory");
}
__device__ __forceinline__ void wait_asynccnt0() {
  asm volatile("s_wait_asynccnt 0x0" ::: "memory");
}

// ---------------- spectral norm: u0 = RandomState(0).randn(D)/||.|| ----------------
__global__ void k_init_u(float* __restrict__ u) {
  if (threadIdx.x != 0 || blockIdx.x != 0) return;
  unsigned mt[624]; int mti;
  mt[0] = 0u;
  for (int i = 1; i < 624; ++i)
    mt[i] = 1812433253u * (mt[i - 1] ^ (mt[i - 1] >> 30)) + (unsigned)i;
  mti = 624;
  auto next = [&]() -> unsigned {
    if (mti >= 624) {
      for (int i = 0; i < 624; ++i) {
        unsigned y = (mt[i] & 0x80000000u) | (mt[(i + 1) % 624] & 0x7fffffffu);
        mt[i] = mt[(i + 397) % 624] ^ (y >> 1) ^ ((y & 1u) ? 0x9908b0dfu : 0u);
      }
      mti = 0;
    }
    unsigned y = mt[mti++];
    y ^= y >> 11; y ^= (y << 7) & 0x9d2c5680u;
    y ^= (y << 15) & 0xefc60000u; y ^= y >> 18;
    return y;
  };
  auto rkdouble = [&]() -> double {
    unsigned a = next() >> 5, b = next() >> 6;
    return (a * 67108864.0 + b) / 9007199254740992.0;
  };
  double cached = 0.0; bool has = false; float ss = 0.f;
  for (int i = 0; i < D; ++i) {
    double g;
    if (has) { g = cached; has = false; }
    else {
      double x1, x2, r2;
      do { x1 = 2.0 * rkdouble() - 1.0; x2 = 2.0 * rkdouble() - 1.0;
           r2 = x1 * x1 + x2 * x2; } while (r2 >= 1.0 || r2 == 0.0);
      double f = sqrt(-2.0 * log(r2) / r2);
      cached = f * x1; has = true; g = f * x2;
    }
    float gf = (float)g;
    u[i] = gf; ss += gf * gf;
  }
  float inv = 1.0f / sqrtf(ss);
  for (int i = 0; i < D; ++i) u[i] *= inv;
}

// ---------------- power iteration (one block, 1024 threads) ----------------
__device__ __forceinline__ float block_sum(float x, float* red) {
  for (int o = 16; o > 0; o >>= 1) x += __shfl_xor(x, o, 32);
  int lane = threadIdx.x & 31, w = threadIdx.x >> 5;
  if (lane == 0) red[w] = x;
  __syncthreads();
  if (w == 0) {
    float r = (lane < 32) ? red[lane] : 0.f;
    for (int o = 16; o > 0; o >>= 1) r += __shfl_xor(r, o, 32);
    if (lane == 0) red[0] = r;
  }
  __syncthreads();
  float r = red[0];
  __syncthreads();
  return r;
}

__global__ void __launch_bounds__(1024) k_power(const float* __restrict__ Wh,
                                                float* __restrict__ u,
                                                float* __restrict__ scale_out) {
  __shared__ float su[D];
  __shared__ float sv[D];
  __shared__ float red[32];
  int j = threadIdx.x;
  su[j] = u[j];
  __syncthreads();
  for (int it = 0; it < 3; ++it) {
    float a = 0.f;                       // v = W^T u
    for (int i = 0; i < D; ++i) a += Wh[(size_t)i * D + j] * su[i];
    float n1 = sqrtf(block_sum(a * a, red));
    sv[j] = a / (n1 + 1e-8f);
    __syncthreads();
    float c = 0.f;                       // u = W v
    for (int k = 0; k < D; ++k) c += Wh[(size_t)j * D + k] * sv[k];
    float n2 = sqrtf(block_sum(c * c, red));
    su[j] = c / (n2 + 1e-8f);
    __syncthreads();
  }
  float wv = 0.f;                        // sigma = |u . (W v)|
  for (int k = 0; k < D; ++k) wv += Wh[(size_t)j * D + k] * sv[k];
  float dot = block_sum(su[j] * wv, red);
  if (j == 0) scale_out[0] = 0.99f / (fabsf(dot) + 1e-8f);
}

// ---------------- convert weights -> bf16, init h ----------------
__global__ void k_convert(const float* __restrict__ Wx, const float* __restrict__ Wh,
                          const float* __restrict__ Wg, const float* __restrict__ h0,
                          const float* __restrict__ scale,
                          u16* __restrict__ wxb, u16* __restrict__ whb,
                          u16* __restrict__ wgb, u16* __restrict__ hbuf,
                          float* __restrict__ hout) {
  int i = blockIdx.x * blockDim.x + threadIdx.x;
  float s = scale[0];
  if (i < D * D) {
    wxb[i] = f2b(Wx[i]);
    whb[i] = f2b(Wh[i] * s);
    wgb[i] = f2b(Wg[i]);
  }
  if (i < Bn * D) {
    hbuf[i] = f2b(h0[i]);
    hout[i] = h0[i];          // h[0] = h0
  }
}

// ---------------- phase A: Xp[t] = x_t @ Wx^T + b  (WMMA, parallel over t) ----------------
__global__ void __launch_bounds__(256) k_xw(const float* __restrict__ x,
                                            const u16* __restrict__ wxb,
                                            const float* __restrict__ b,
                                            float* __restrict__ xp) {
  __shared__ u16 xs[Bn * D];            // 32 KB, x_t in bf16
  int t = blockIdx.x;
  const float* xt = x + (size_t)t * Bn * D;
  for (int i = threadIdx.x; i < Bn * D; i += 256) xs[i] = f2b(xt[i]);
  __syncthreads();

  int lane = threadIdx.x & 31;
  int wave = threadIdx.x >> 5;
  int nbase = wave * 128;               // 8 waves x 128 cols = 1024
  v8f acc[8] = {};
  for (int ks = 0; ks < D / 32; ++ks) {
    int kb = ks * 32;
    v16bf a = load_frag_A(xs, lane, kb, D);
    v16bf bm[8];
#pragma unroll
    for (int j = 0; j < 8; ++j)
      bm[j] = load_frag_B(wxb, lane, nbase + j * 16, kb, D);
#pragma unroll
    for (int j = 0; j < 8; ++j)
      acc[j] = __builtin_amdgcn_wmma_f32_16x16x32_bf16(
          false, a, false, bm[j], (short)0, acc[j], false, false);
  }
  int m0 = (lane >> 4) * 8;
#pragma unroll
  for (int j = 0; j < 8; ++j) {
    int n = nbase + j * 16 + (lane & 15);
    float bias = b[n];
    float* op = xp + (size_t)t * Bn * D + n;
#pragma unroll
    for (int v = 0; v < 8; ++v)
      op[(size_t)(m0 + v) * D] = acc[j][v] + bias;
  }
}

// ---------------- phase B: persistent gated scan ----------------
__global__ void __launch_bounds__(SCAN_THREADS) k_scan(
    const float* __restrict__ z,
    const u16* __restrict__ whb, const u16* __restrict__ wgb,
    u16* __restrict__ hbuf,              // 2 * Bn*D bf16 ping-pong
    float* __restrict__ outs,            // d_out (holds Xp, overwritten with outs)
    float* __restrict__ hout,            // d_out + T*Bn*D
    unsigned* __restrict__ cnt) {
  extern __shared__ char smem_raw[];
  u16* whs = (u16*)smem_raw;             // 4 waves * 16 rows * D   (128 KB)
  u16* wgs = whs + 4 * 16 * D;           // 128 KB
  u16* hs  = wgs + 4 * 16 * D;           // 32 KB : staged h (Bn x D)
  // LDS byte offset of hs: dynamic LDS begins after static allocation
  const unsigned hs_lds_base =
      __builtin_amdgcn_groupstaticsize() + (unsigned)(8 * 16 * D * 2);

  int lane = threadIdx.x & 31;
  int wave = threadIdx.x >> 5;
  int nt = blockIdx.x * 4 + wave;        // 64 N-tiles total = D/16
  int n0 = nt * 16;

  // stage this wave's weight slices (rows n0..n0+15) into LDS once
  {
    const uint4* swh = (const uint4*)(whb + (size_t)n0 * D);
    const uint4* swg = (const uint4*)(wgb + (size_t)n0 * D);
    uint4* dwh = (uint4*)(whs + wave * 16 * D);
    uint4* dwg = (uint4*)(wgs + wave * 16 * D);
    for (int i = lane; i < 16 * D / 8; i += 32) { dwh[i] = swh[i]; dwg[i] = swg[i]; }
  }
  // stage h0 via async LDS DMA
  for (int i = threadIdx.x; i < Bn * D / 8; i += SCAN_THREADS)
    async_b128_to_lds(hs_lds_base + (unsigned)i * 16u,
                      (const char*)hbuf + (size_t)i * 16u);
  wait_asynccnt0();
  __syncthreads();

  const u16* whw = whs + wave * 16 * D;  // local rows 0..15 = global n0..n0+15
  const u16* wgw = wgs + wave * 16 * D;
  int m0 = (lane >> 4) * 8;
  int n  = n0 + (lane & 15);

  for (int t = 0; t < Tn; ++t) {
    // prefetch next step's fresh HBM lines while we compute
    if (t + 1 < Tn) {
      __builtin_prefetch(outs + (size_t)(t + 1) * Bn * D + n, 0, 0);
      __builtin_prefetch(z + (size_t)(t + 1) * Bn * D + n, 0, 0);
    }

    // ---- h_new = tanh(Xp[t] + h_prev @ Wh^T) ----  (2 interleaved acc chains)
    v8f acc0 = {}, acc1 = {};
    for (int ks = 0; ks < D / 32; ks += 2) {
      v16bf a0 = load_frag_A(hs, lane, ks * 32, D);
      v16bf b0 = load_frag_B(whw, lane, 0, ks * 32, D);
      v16bf a1 = load_frag_A(hs, lane, ks * 32 + 32, D);
      v16bf b1 = load_frag_B(whw, lane, 0, ks * 32 + 32, D);
      acc0 = __builtin_amdgcn_wmma_f32_16x16x32_bf16(
          false, a0, false, b0, (short)0, acc0, false, false);
      acc1 = __builtin_amdgcn_wmma_f32_16x16x32_bf16(
          false, a1, false, b1, (short)0, acc1, false, false);
    }
    v8f acc = acc0 + acc1;

    float hn[8];
    const float* xpt = outs + (size_t)t * Bn * D + n;
    float* hot = hout + (size_t)(t + 1) * Bn * D + n;
    u16*   hbn = hbuf + (size_t)((t + 1) & 1) * Bn * D + n;
#pragma unroll
    for (int v = 0; v < 8; ++v) {
      float val = tanhf(acc[v] + xpt[(size_t)(m0 + v) * D]);
      hn[v] = val;
      hot[(size_t)(m0 + v) * D] = val;
      hbn[(size_t)(m0 + v) * D] = f2b(val);
    }

    // ---- grid-wide barrier ----
    __threadfence();
    __syncthreads();
    if (threadIdx.x == 0) {
      atomicAdd(cnt, 1u);
      unsigned target = (unsigned)(t + 1) * NBLK;
      while (atomicAdd(cnt, 0u) < target) __builtin_amdgcn_s_sleep(1);
      __threadfence();
    }
    __syncthreads();

    // stage h_new for gate GEMM and next step (async LDS DMA)
    {
      const char* hb = (const char*)(hbuf + (size_t)((t + 1) & 1) * Bn * D);
      for (int i = threadIdx.x; i < Bn * D / 8; i += SCAN_THREADS)
        async_b128_to_lds(hs_lds_base + (unsigned)i * 16u, hb + (size_t)i * 16u);
      wait_asynccnt0();
    }
    __syncthreads();

    // ---- out = h_new * sigmoid(z_t + h_new @ Wg^T) ----
    v8f g0 = {}, g1 = {};
    for (int ks = 0; ks < D / 32; ks += 2) {
      v16bf a0 = load_frag_A(hs, lane, ks * 32, D);
      v16bf b0 = load_frag_B(wgw, lane, 0, ks * 32, D);
      v16bf a1 = load_frag_A(hs, lane, ks * 32 + 32, D);
      v16bf b1 = load_frag_B(wgw, lane, 0, ks * 32 + 32, D);
      g0 = __builtin_amdgcn_wmma_f32_16x16x32_bf16(
          false, a0, false, b0, (short)0, g0, false, false);
      g1 = __builtin_amdgcn_wmma_f32_16x16x32_bf16(
          false, a1, false, b1, (short)0, g1, false, false);
    }
    v8f acc2 = g0 + g1;

    const float* zt = z + (size_t)t * Bn * D + n;
    float* op = outs + (size_t)t * Bn * D + n;
#pragma unroll
    for (int v = 0; v < 8; ++v) {
      float g = 1.0f / (1.0f + expf(-(zt[(size_t)(m0 + v) * D] + acc2[v])));
      op[(size_t)(m0 + v) * D] = hn[v] * g;
    }
  }
}

// ---------------- host ----------------
extern "C" void kernel_launch(void* const* d_in, const int* in_sizes, int n_in,
                              void* d_out, int out_size, void* d_ws, size_t ws_size,
                              hipStream_t stream) {
  const float* x  = (const float*)d_in[0];
  const float* z  = (const float*)d_in[1];
  const float* h0 = (const float*)d_in[2];
  const float* Wx = (const float*)d_in[3];
  const float* Wh = (const float*)d_in[4];
  const float* Wg = (const float*)d_in[5];
  const float* b  = (const float*)d_in[6];
  float* out = (float*)d_out;

  char* ws = (char*)d_ws;
  float*    u     = (float*)(ws + 0);
  float*    scale = (float*)(ws + 8192);
  unsigned* cnt   = (unsigned*)(ws + 12288);
  u16* wxb  = (u16*)(ws + 16384);
  u16* whb  = wxb + (size_t)D * D;
  u16* wgb  = whb + (size_t)D * D;
  u16* hbuf = wgb + (size_t)D * D;        // 2 * Bn*D bf16

  float* outs = out;                       // [T,B,D] (Xp staging then outs)
  float* hout = out + (size_t)Tn * Bn * D; // [T+1,B,D]

  hipMemsetAsync(cnt, 0, sizeof(unsigned), stream);
  k_init_u<<<1, 1, 0, stream>>>(u);
  k_power<<<1, 1024, 0, stream>>>(Wh, u, scale);
  k_convert<<<(D * D + 255) / 256, 256, 0, stream>>>(Wx, Wh, Wg, h0, scale,
                                                     wxb, whb, wgb, hbuf, hout);
  k_xw<<<Tn, 256, 0, stream>>>(x, wxb, b, outs);
  size_t smem = (size_t)(8 * 16 * D + Bn * D) * sizeof(u16);  // 288 KB
  k_scan<<<NBLK, SCAN_THREADS, smem, stream>>>(z, whb, wgb, hbuf, outs, hout, cnt);
}
